// PairwiseAttentionBlock_46420006535528
// MI455X (gfx1250) — matrix operand order repaired and checked by reading, not compiled, not run on hardware
//
#include <hip/hip_runtime.h>
#include <hip/hip_bf16.h>

// PairwiseAttentionBlock for MI455X (gfx1250), bf16 WMMA path.
// DIM=128, HEADS=4, DIM_HEAD=32, N=320, B=1.
// ~67 GFLOP, ~52MB per [N,N,D] tensor -> bandwidth-leaning; all matmuls via
// v_wmma_f32_16x16x32_bf16 with f32 accumulation, bf16 intermediates.

#define NSEQ   320
#define NN     (NSEQ * NSEQ)        // 102400 positions
#define DIMD   128
#define HEADS  4
#define DHEAD  32
#define INNER  128
#define QK_SCALE 0.17677669529663687f  // 32^-0.5

typedef __bf16 bf16_t;
typedef __attribute__((ext_vector_type(16))) __bf16 v16bf;
typedef __attribute__((ext_vector_type(8)))  __bf16 v8bf;
typedef __attribute__((ext_vector_type(8)))  float  v8f;

__device__ __forceinline__ v8f wmma_bf16(v16bf a, v16bf b, v8f c) {
  return __builtin_amdgcn_wmma_f32_16x16x32_bf16(false, a, false, b, (short)0,
                                                 c, false, false);
}

// A-fragment (16M x 32K bf16) from row-major [rows x stride]:
// lane l<16: row=row0+l, K = k0+8h+{0..7} and k0+16+8h+{0..7}, h = (lane>=16)
__device__ __forceinline__ v16bf load_fragA(const bf16_t* base, int stride,
                                            int row0, int k0) {
  int lane = threadIdx.x & 31;
  int half = lane >> 4, li = lane & 15;
  const bf16_t* p = base + (row0 + li) * stride + k0 + 8 * half;
  v8bf lo = *(const v8bf*)p;
  v8bf hi = *(const v8bf*)(p + 16);
  v16bf f;
#pragma unroll
  for (int i = 0; i < 8; ++i) { f[i] = lo[i]; f[i + 8] = hi[i]; }
  return f;
}

// B-fragment (32K x 16N bf16) given rows of B^T (row-major [N x stride]):
// lane l: N=row0+(l&15), K = k0+8*half+{0..7} and k0+16+8*half+{0..7}
__device__ __forceinline__ v16bf load_fragB(const bf16_t* base, int stride,
                                            int row0, int k0) {
  int lane = threadIdx.x & 31;
  int half = lane >> 4, li = lane & 15;
  const bf16_t* p = base + (row0 + li) * stride + k0 + 8 * half;
  v8bf lo = *(const v8bf*)p;
  v8bf hi = *(const v8bf*)(p + 16);
  v16bf f;
#pragma unroll
  for (int i = 0; i < 8; ++i) { f[i] = lo[i]; f[i + 8] = hi[i]; }
  return f;
}

// ---------------------------------------------------------------- weights ---
// wcat (bf16 [512 x 128]) = rows of [wq;wkv;wg]; wob (bf16 [128 x 128]) = wo.
__global__ __launch_bounds__(256) void wconv_kernel(
    const float* __restrict__ wq, const float* __restrict__ wkv,
    const float* __restrict__ wg, const float* __restrict__ wo,
    bf16_t* __restrict__ wcat, bf16_t* __restrict__ wob) {
  int t = blockIdx.x * 256 + threadIdx.x;
  if (t < 128 * 128)            wcat[t] = (bf16_t)wq[t];
  else if (t < 384 * 128)       wcat[t] = (bf16_t)wkv[t - 128 * 128];
  else if (t < 512 * 128)       wcat[t] = (bf16_t)wg[t - 384 * 128];
  int u = t - 512 * 128;
  if (u >= 0 && u < 128 * 128)  wob[u] = (bf16_t)wo[u];
}

// ------------------------------------------------------- layernorm + bias ---
// One block (128 thr) per position (a,b). Writes xn (bf16, transposed for the
// column pass) and biasT[h][j][i] = x[i,j]·wb[h]  (j-major for coalesced
// access inside the S^T attention tiles).
__global__ __launch_bounds__(128) void ln_bias_kernel(
    const float* __restrict__ x, const float* __restrict__ ng,
    const float* __restrict__ nb, const float* __restrict__ wb,
    bf16_t* __restrict__ xn, float* __restrict__ biasT, int transpose) {
  __shared__ float red[128];
  int pos = blockIdx.x;
  int d = threadIdx.x;
  int a = pos / NSEQ, b = pos - a * NSEQ;
  float v = x[pos * DIMD + d];

  red[d] = v; __syncthreads();
#pragma unroll
  for (int off = 64; off > 0; off >>= 1) {
    if (d < off) red[d] += red[d + off];
    __syncthreads();
  }
  float mu = red[0] * (1.0f / DIMD);
  __syncthreads();
  float diff = v - mu;
  red[d] = diff * diff; __syncthreads();
#pragma unroll
  for (int off = 64; off > 0; off >>= 1) {
    if (d < off) red[d] += red[d + off];
    __syncthreads();
  }
  float var = red[0] * (1.0f / DIMD);
  __syncthreads();
  float normed = diff * rsqrtf(var + 1e-5f) * ng[d] + nb[d];

  int dst = transpose ? (b * NSEQ + a) : pos;
  xn[dst * DIMD + d] = (bf16_t)normed;

  // bias projections (4 dots of 128) from un-normalized x
  for (int h = 0; h < HEADS; ++h) {
    red[d] = v * wb[h * DIMD + d]; __syncthreads();
#pragma unroll
    for (int off = 64; off > 0; off >>= 1) {
      if (d < off) red[d] += red[d + off];
      __syncthreads();
    }
    if (d == 0) biasT[(h * NSEQ + b) * NSEQ + a] = red[0];
    __syncthreads();
  }
}

// --------------------------------------------------- fused qkv/gate GEMM ----
// Y[102400 x 512] = Xn[102400 x 128] x Wcat^T ; epilogue scatters into
// qb (scaled), kb, vt (V transposed per (r,h): [32 x 320]) and sigmoid gates.
__global__ __launch_bounds__(32) void proj_gemm_kernel(
    const bf16_t* __restrict__ xn, const bf16_t* __restrict__ wcat,
    const float* __restrict__ bg, bf16_t* __restrict__ qb,
    bf16_t* __restrict__ kb, bf16_t* __restrict__ vt,
    bf16_t* __restrict__ gates) {
  int m0 = blockIdx.x * 16;
  int n0 = blockIdx.y * 64;
  int lane = threadIdx.x;
  int half = lane >> 4, li = lane & 15;

  v8f acc[4] = {v8f{}, v8f{}, v8f{}, v8f{}};
#pragma unroll
  for (int k0 = 0; k0 < DIMD; k0 += 32) {
    v16bf a = load_fragA(xn, DIMD, m0, k0);
#pragma unroll
    for (int t = 0; t < 4; ++t) {
      v16bf b = load_fragB(wcat, DIMD, n0 + 16 * t, k0);
      acc[t] = wmma_bf16(a, b, acc[t]);
    }
  }
#pragma unroll
  for (int t = 0; t < 4; ++t)
#pragma unroll
    for (int r = 0; r < 8; ++r) {
      int row = m0 + r + 8 * half;
      int col = n0 + 16 * t + li;
      float v = acc[t][r];
      if (col < 128) {                     // q (pre-scaled)
        qb[row * INNER + col] = (bf16_t)(v * QK_SCALE);
      } else if (col < 256) {              // k
        kb[row * INNER + (col - 128)] = (bf16_t)v;
      } else if (col < 384) {              // v -> transposed [r][h][e][l]
        int c = col - 256, h = c >> 5, e = c & 31;
        int ra = row / NSEQ, la = row - ra * NSEQ;
        vt[(((ra * HEADS + h) << 5) + e) * NSEQ + la] = (bf16_t)v;
      } else {                             // gate = sigmoid(v + bg)
        int c = col - 384;
        float g = 1.0f / (1.0f + __expf(-(v + bg[c])));
        gates[row * INNER + c] = (bf16_t)g;
      }
    }
}

// ------------------------------------------------ flash attention (WMMA) ----
// One wave per (axial row r, head h, i-tile of 16). Computes S^T = K·Q^T so
// softmax state is per-lane (lane = i); exp probs feed the attn·V WMMA as the
// A-fragment with no cross-lane shuffle of matrix data.
__global__ __launch_bounds__(128) void attn_kernel(
    const bf16_t* __restrict__ qb, const bf16_t* __restrict__ kb,
    const bf16_t* __restrict__ vt, const float* __restrict__ biasT,
    bf16_t* __restrict__ attnout) {
  int lane = threadIdx.x;            // 32
  int h    = threadIdx.y;            // 4 waves / block, one head each
  int r_ax = blockIdx.x;             // 320
  int ib   = blockIdx.y * 16;        // 20 i-tiles
  int half = lane >> 4, li = lane & 15;

  const bf16_t* qbase = qb + (r_ax * NSEQ) * INNER + h * DHEAD;
  const bf16_t* kbase = kb + (r_ax * NSEQ) * INNER + h * DHEAD;
  const bf16_t* vbase = vt + (r_ax * HEADS + h) * DHEAD * NSEQ;
  const float*  bbase = biasT + h * NSEQ * NSEQ;   // [j][i]

  v16bf qf = load_fragB(qbase, INNER, ib, 0);      // Q as B-frag, reused
  v8f o0 = v8f{}, o1 = v8f{};
  float m = -1e30f, s = 0.0f;

  for (int jb = 0; jb < NSEQ; jb += 32) {
    // S^T tiles: A = K rows (j), B = Q^T, K-dim = dhead 32 (one WMMA each)
    v16bf ka0 = load_fragA(kbase, INNER, jb, 0);
    v16bf ka1 = load_fragA(kbase, INNER, jb + 16, 0);
    v8f s0 = wmma_bf16(ka0, qf, v8f{});
    v8f s1 = wmma_bf16(ka1, qf, v8f{});

    float vals[16];
#pragma unroll
    for (int r = 0; r < 8; ++r) {
      vals[r]     = s0[r] + bbase[(jb      + r + 8 * half) * NSEQ + ib + li];
      vals[r + 8] = s1[r] + bbase[(jb + 16 + r + 8 * half) * NSEQ + ib + li];
    }
    float cm = vals[0];
#pragma unroll
    for (int u = 1; u < 16; ++u) cm = fmaxf(cm, vals[u]);
    cm = fmaxf(cm, __shfl_xor(cm, 16));    // merge j-halves (lanes l, l^16 share i)
    float mn = fmaxf(m, cm);
    float sc = __expf(m - mn);
    float psum = 0.0f;
    v16bf af;                               // exp probs already in A-layout
#pragma unroll
    for (int u = 0; u < 16; ++u) {
      float p = __expf(vals[u] - mn);
      psum += p;
      af[u] = (bf16_t)p;
    }
    s = s * sc + psum;
    m = mn;
#pragma unroll
    for (int r = 0; r < 8; ++r) {           // rescale O rows (i = r + 8*half)
      float f = __shfl(sc, r + 8 * half);
      o0[r] *= f; o1[r] *= f;
    }
    v16bf vb0 = load_fragB(vbase, NSEQ, 0,  jb);   // V^T rows e=0..15
    v16bf vb1 = load_fragB(vbase, NSEQ, 16, jb);   // e=16..31
    o0 = wmma_bf16(af, vb0, o0);
    o1 = wmma_bf16(af, vb1, o1);
  }

  s += __shfl_xor(s, 16);
  float inv = 1.0f / s;
  bf16_t* obase = attnout + (r_ax * NSEQ + ib) * INNER + h * DHEAD;
#pragma unroll
  for (int r = 0; r < 8; ++r) {
    float f = __shfl(inv, r + 8 * half);
    int ro = (r + 8 * half) * INNER;
    obase[ro + li]      = (bf16_t)(o0[r] * f);
    obase[ro + 16 + li] = (bf16_t)(o1[r] * f);
  }
}

// ------------------------------------- gated output projection + residual ---
// Z = (attnout*gates)[102400x128] x wo^T + bo + residual; un-transposes for
// the column pass on writeback.
__global__ __launch_bounds__(32) void out_proj_kernel(
    const bf16_t* __restrict__ attnout, const bf16_t* __restrict__ gates,
    const bf16_t* __restrict__ wob, const float* __restrict__ bo,
    const float* __restrict__ xsrc, float* __restrict__ xdst, int transpose) {
  int m0 = blockIdx.x * 16;
  int n0 = blockIdx.y * 64;
  int lane = threadIdx.x;
  int half = lane >> 4, li = lane & 15;

  v8f acc[4] = {v8f{}, v8f{}, v8f{}, v8f{}};
#pragma unroll
  for (int k0 = 0; k0 < INNER; k0 += 32) {
    int row = m0 + li;
    const bf16_t* pa = attnout + row * INNER + k0 + 8 * half;
    const bf16_t* pg = gates   + row * INNER + k0 + 8 * half;
    v8bf a_lo = *(const v8bf*)pa, a_hi = *(const v8bf*)(pa + 16);
    v8bf g_lo = *(const v8bf*)pg, g_hi = *(const v8bf*)(pg + 16);
    v16bf a;
#pragma unroll
    for (int i = 0; i < 8; ++i) {
      a[i]     = (bf16_t)((float)a_lo[i] * (float)g_lo[i]);
      a[i + 8] = (bf16_t)((float)a_hi[i] * (float)g_hi[i]);
    }
#pragma unroll
    for (int t = 0; t < 4; ++t) {
      v16bf b = load_fragB(wob, INNER, n0 + 16 * t, k0);
      acc[t] = wmma_bf16(a, b, acc[t]);
    }
  }
#pragma unroll
  for (int t = 0; t < 4; ++t)
#pragma unroll
    for (int r = 0; r < 8; ++r) {
      int row = m0 + r + 8 * half;              // xt-space position
      int d = n0 + 16 * t + li;
      int ra = row / NSEQ, la = row - ra * NSEQ;
      int dst = transpose ? (la * NSEQ + ra) : row;
      xdst[dst * DIMD + d] = acc[t][r] + bo[d] + xsrc[dst * DIMD + d];
    }
}

// ---------------------------------------------------------------- launch ----
extern "C" void kernel_launch(void* const* d_in, const int* in_sizes, int n_in,
                              void* d_out, int out_size, void* d_ws,
                              size_t ws_size, hipStream_t stream) {
  (void)in_sizes; (void)n_in; (void)out_size; (void)ws_size;
  const float* x_in = (const float*)d_in[0];
  // per-pass weights: ng, nb, wb, wq, wkv, wg, bg, wo, bo
  const float* W[2][9];
  for (int p = 0; p < 2; ++p)
    for (int q = 0; q < 9; ++q)
      W[p][q] = (const float*)d_in[1 + 9 * p + q];

  char* ws = (char*)d_ws;
  size_t off = 0;
  auto alloc = [&](size_t bytes) -> void* {
    void* p = ws + off;
    off = (off + bytes + 255) & ~(size_t)255;
    return p;
  };
  float*  xcur    = (float*)alloc((size_t)NN * DIMD * 4);   // pass-1 output
  bf16_t* xn      = (bf16_t*)alloc((size_t)NN * DIMD * 2);
  bf16_t* qb      = (bf16_t*)alloc((size_t)NN * INNER * 2);
  bf16_t* kb      = (bf16_t*)alloc((size_t)NN * INNER * 2);
  bf16_t* vtb     = (bf16_t*)alloc((size_t)NN * INNER * 2);
  bf16_t* gates   = (bf16_t*)alloc((size_t)NN * INNER * 2);
  bf16_t* attnout = (bf16_t*)alloc((size_t)NN * INNER * 2);
  float*  biasT   = (float*)alloc((size_t)HEADS * NN * 4);
  bf16_t* wcat[2], *wob[2];
  for (int p = 0; p < 2; ++p) {
    wcat[p] = (bf16_t*)alloc(512 * 128 * 2);
    wob[p]  = (bf16_t*)alloc(128 * 128 * 2);
  }

  for (int p = 0; p < 2; ++p)
    wconv_kernel<<<(512 * 128 + 128 * 128 + 255) / 256, 256, 0, stream>>>(
        W[p][3], W[p][4], W[p][5], W[p][7], wcat[p], wob[p]);

  auto run_pass = [&](const float* xin, float* xout, int p, int transpose) {
    ln_bias_kernel<<<NN, 128, 0, stream>>>(xin, W[p][0], W[p][1], W[p][2], xn,
                                           biasT, transpose);
    proj_gemm_kernel<<<dim3(NN / 16, 8), 32, 0, stream>>>(
        xn, wcat[p], W[p][6], qb, kb, vtb, gates);
    attn_kernel<<<dim3(NSEQ, NSEQ / 16), dim3(32, 4), 0, stream>>>(
        qb, kb, vtb, biasT, attnout);
    out_proj_kernel<<<dim3(NN / 16, 2), 32, 0, stream>>>(
        attnout, gates, wob[p], W[p][8], xin, xout, transpose);
  };

  run_pass(x_in, xcur, 0, /*transpose=*/0);           // row-wise axial
  run_pass(xcur, (float*)d_out, 1, /*transpose=*/1);  // column-wise axial
}
